// AAGA_65644280152705
// MI455X (gfx1250) — matrix operand order repaired and checked by reading, not compiled, not run on hardware
//
#include <hip/hip_runtime.h>
#include <hip/hip_bf16.h>

// ---------------------------------------------------------------------------
// Agent-attention block for MI455X (gfx1250), wave32 + WMMA bf16 pipeline.
// B=8, N=8192, D=256, K=64 agents. All GEMM-shaped ops use
// v_wmma_f32_16x16x32_bf16 (f32 accumulate). Intermediates stored bf16.
// Stage-1 K-chunk staging uses the Tensor Data Mover (tensor_load_to_lds +
// s_wait_tensorcnt); all WMMA operands load as 16-byte contiguous vectors.
// ---------------------------------------------------------------------------

typedef __attribute__((ext_vector_type(16))) __bf16 v16bf;
typedef __attribute__((ext_vector_type(8)))  __bf16 v8bf;
typedef __attribute__((ext_vector_type(8)))  float  v8f;
typedef __attribute__((ext_vector_type(4)))  float  f32x4;
typedef __attribute__((ext_vector_type(4)))  unsigned int u32x4;
typedef __attribute__((ext_vector_type(8)))  int    i32x8;
typedef __attribute__((ext_vector_type(4)))  int    i32x4;

#define BATCH 8
#define NTOK  8192
#define DIM   256
#define KAG   64
#define CHUNK 64                  // stage-1 token chunk
#define NCH   (NTOK / CHUNK)      // 128 chunks per batch
#define SCALE 0.0625f             // 256^-0.5

__device__ inline v8f vzero() {
  v8f z;
#pragma unroll
  for (int i = 0; i < 8; ++i) z[i] = 0.0f;
  return z;
}

__device__ inline v8f wmma_bf16(v16bf a, v16bf b, v8f c) {
  return __builtin_amdgcn_wmma_f32_16x16x32_bf16(
      /*neg_a=*/false, a, /*neg_b=*/false, b,
      /*c_mod=*/(short)0, c, /*reuse_a=*/false, /*reuse_b=*/false);
}

// TDM: DMA `nelem2b` 2-byte elements from contiguous global memory into LDS.
// Issued by the calling wave (EXEC ignored); completion via TENSORcnt.
__device__ inline void tdm_load_to_lds(const void* gsrc, void* ldst,
                                       unsigned nelem2b) {
  unsigned ldsa = (unsigned)(size_t)ldst;          // LDS byte offset
  unsigned long long ga = (unsigned long long)(size_t)gsrc;
  u32x4 g0;
  g0[0] = 1u;                                      // count=1, user descriptor
  g0[1] = ldsa;                                    // lds_addr
  g0[2] = (unsigned)(ga & 0xffffffffu);            // global_addr[31:0]
  g0[3] = (unsigned)((ga >> 32) & 0x1ffffffu)      // global_addr[56:32]
        | (2u << 30);                              // type = 2 ("image")
  i32x8 g1;
  g1[0] = 1 << 16;                                 // data_size = 2 bytes
  g1[1] = (int)(nelem2b << 16);                    // tensor_dim0[15:0]
  g1[2] = (int)((nelem2b >> 16) & 0xffffu) | (1 << 16); // dim0 hi | tensor_dim1=1
  g1[3] = (int)(nelem2b << 16);                    // tile_dim0
  g1[4] = 1;                                       // tile_dim1 = 1
  g1[5] = (int)nelem2b;                            // tensor_dim0_stride
  g1[6] = 0;
  g1[7] = 0;
  i32x4 gz = {0, 0, 0, 0};
#if __clang_major__ >= 23
  i32x8 gz8 = {0, 0, 0, 0, 0, 0, 0, 0};
  __builtin_amdgcn_tensor_load_to_lds(g0, g1, gz, gz, gz8, 0);
#else
  __builtin_amdgcn_tensor_load_to_lds(g0, g1, gz, gz, 0);
#endif
}

// ---- A fragment loaders: row-major M x ld, K contiguous. -------------------
__device__ inline v16bf ld_a_bf(const __bf16* p, int ld, int row0, int k0) {
  int lane = threadIdx.x & 31, hi = lane >> 4, r = lane & 15;
  const __bf16* base = p + (size_t)(row0 + r) * ld + k0 + hi * 8;
  v8bf lo = *(const v8bf*)(base);
  v8bf hh = *(const v8bf*)(base + 16);
  v16bf f;
#pragma unroll
  for (int e = 0; e < 8; ++e) { f[e] = lo[e]; f[e + 8] = hh[e]; }
  return f;
}

__device__ inline v16bf ld_a_f32(const float* p, int ld, int row0, int k0) {
  int lane = threadIdx.x & 31, hi = lane >> 4, r = lane & 15;
  const float* base = p + (size_t)(row0 + r) * ld + k0 + hi * 8;
  f32x4 a0 = *(const f32x4*)(base);
  f32x4 a1 = *(const f32x4*)(base + 4);
  f32x4 b0 = *(const f32x4*)(base + 16);
  f32x4 b1 = *(const f32x4*)(base + 20);
  v16bf f;
#pragma unroll
  for (int e = 0; e < 4; ++e) {
    f[e]      = (__bf16)a0[e];
    f[e + 4]  = (__bf16)a1[e];
    f[e + 8]  = (__bf16)b0[e];
    f[e + 12] = (__bf16)b1[e];
  }
  return f;
}

// ---- B fragment loaders for C = A * B^T (second operand row-major [n][k]).
__device__ inline v16bf ld_bt_bf(const __bf16* p, int ld, int n0, int k0) {
  int lane = threadIdx.x & 31, hi = lane >> 4, n = lane & 15;
  const __bf16* base = p + (size_t)(n0 + n) * ld + k0 + hi * 8;
  v8bf lo = *(const v8bf*)(base);
  v8bf hh = *(const v8bf*)(base + 16);
  v16bf f;
#pragma unroll
  for (int e = 0; e < 8; ++e) { f[e] = lo[e]; f[e + 8] = hh[e]; }
  return f;
}

__device__ inline v16bf ld_bt_f32(const float* p, int ld, int n0, int k0) {
  int lane = threadIdx.x & 31, hi = lane >> 4, n = lane & 15;
  const float* base = p + (size_t)(n0 + n) * ld + k0 + hi * 8;
  v16bf f;
#pragma unroll
  for (int q = 0; q < 4; ++q) {
    f32x4 t = *(const f32x4*)(base + (q & 1) * 4 + (q >> 1) * 16);
#pragma unroll
    for (int e = 0; e < 4; ++e) f[q * 4 + e] = (__bf16)t[e];
  }
  return f;
}

// C/D 16x16 f32: lane<16 col=lane rows 0..7; lane>=16 col=lane-16 rows 8..15.
__device__ inline void st_c_f32(float* p, int ld, int m0, int n0, v8f c, float s) {
  int lane = threadIdx.x & 31, hi = lane >> 4, n = lane & 15;
#pragma unroll
  for (int r = 0; r < 8; ++r) p[(size_t)(m0 + r + hi * 8) * ld + n0 + n] = c[r] * s;
}

__device__ inline void st_c_bf16(__bf16* p, int ld, int m0, int n0, v8f c) {
  int lane = threadIdx.x & 31, hi = lane >> 4, n = lane & 15;
#pragma unroll
  for (int r = 0; r < 8; ++r)
    p[(size_t)(m0 + r + hi * 8) * ld + n0 + n] = (__bf16)c[r];
}

// ---------------------------------------------------------------------------
// K1: qa_ka = agent @ W_agent^T + b_agent  -> q_agent, k_agent (bf16)
// ---------------------------------------------------------------------------
__global__ void agent_proj_kernel(const float* __restrict__ agent,
                                  const float* __restrict__ Wag,
                                  const float* __restrict__ bag,
                                  __bf16* __restrict__ qa,
                                  __bf16* __restrict__ ka) {
  int tid = threadIdx.x, wave = tid >> 5;
  for (int t = wave; t < 128; t += 8) {        // 4 m-tiles x 32 n-tiles
    int mt = t & 3, nt = t >> 2;
    v8f acc = vzero();
    for (int ks = 0; ks < DIM; ks += 32) {
      v16bf a = ld_a_f32(agent, DIM, mt * 16, ks);
      v16bf b = ld_bt_f32(Wag, DIM, nt * 16, ks);
      acc = wmma_bf16(a, b, acc);
    }
    int lane = tid & 31, hi = lane >> 4, nl = lane & 15;
    int col = nt * 16 + nl;
    float bias = bag[col];
    __bf16* dstp = (col < DIM) ? qa : ka;
    int c2 = (col < DIM) ? col : col - DIM;
#pragma unroll
    for (int r = 0; r < 8; ++r) {
      int row = mt * 16 + r + hi * 8;
      dstp[row * DIM + c2] = (__bf16)(acc[r] + bias);
    }
  }
}

// ---------------------------------------------------------------------------
// K2: qkv = x @ W_qkv^T + b_qkv, stored bf16 into q/k/v.
// ---------------------------------------------------------------------------
__global__ void qkv_kernel(const float* __restrict__ x,
                           const float* __restrict__ Wqkv,
                           const float* __restrict__ bqkv,
                           __bf16* __restrict__ q, __bf16* __restrict__ k,
                           __bf16* __restrict__ v) {
  __shared__ __bf16 sW[64 * DIM];              // 32 KB: this block's W slice
  int ntile = blockIdx.y;                      // 0..11
  int mbase = blockIdx.x * 128;
  int tid = threadIdx.x, wave = tid >> 5;

  const float* Wb = Wqkv + (size_t)ntile * 64 * DIM;
  {
    const f32x4* src = (const f32x4*)Wb;
    for (int i = tid; i < 64 * DIM / 4; i += 256) {
      f32x4 t = src[i];
      __bf16* d = sW + i * 4;
      d[0] = (__bf16)t[0]; d[1] = (__bf16)t[1];
      d[2] = (__bf16)t[2]; d[3] = (__bf16)t[3];
    }
  }
  __syncthreads();

  int mrow = mbase + wave * 16;
  v8f acc[4];
#pragma unroll
  for (int i = 0; i < 4; ++i) acc[i] = vzero();

  for (int ks = 0; ks < DIM; ks += 32) {
    v16bf a = ld_a_f32(x, DIM, mrow, ks);
#pragma unroll
    for (int nt = 0; nt < 4; ++nt) {
      v16bf b = ld_bt_bf(sW, DIM, nt * 16, ks);
      acc[nt] = wmma_bf16(a, b, acc[nt]);
    }
  }

  int lane = tid & 31, hi = lane >> 4, nl = lane & 15;
  int which = ntile >> 2;                      // 0=q 1=k 2=v
  int col0 = (ntile & 3) * 64;
  __bf16* dst = (which == 0) ? q : (which == 1) ? k : v;
#pragma unroll
  for (int nt = 0; nt < 4; ++nt) {
    int col = col0 + nt * 16 + nl;
    float bias = bqkv[ntile * 64 + nt * 16 + nl];
#pragma unroll
    for (int r = 0; r < 8; ++r) {
      size_t row = (size_t)mrow + r + hi * 8;
      dst[row * DIM + col] = (__bf16)(acc[nt][r] + bias);
    }
  }
}

// ---------------------------------------------------------------------------
// K3: stage-1 flash chunk: S1 = Qa @ Kc^T * sc, local softmax stats,
// acc = P1 @ Vc. K chunk staged via the Tensor Data Mover; V staged
// TRANSPOSED so P@V B-operand loads are contiguous.
// ---------------------------------------------------------------------------
__global__ void stage1_kernel(const __bf16* __restrict__ qa,
                              const __bf16* __restrict__ kb,
                              const __bf16* __restrict__ vb,
                              float* __restrict__ part_m,
                              float* __restrict__ part_l,
                              float* __restrict__ part_acc) {
  int chunk = blockIdx.x, b = blockIdx.y;
  size_t rowbase = (size_t)b * NTOK + (size_t)chunk * CHUNK;
  const __bf16* kbase = kb + rowbase * DIM;
  const __bf16* vbase = vb + rowbase * DIM;

  __shared__ __bf16 sKV[CHUNK * DIM];          // 32 KB: K (then V^T)
  __shared__ float  sS[KAG * CHUNK];           // 16 KB logits
  __shared__ __bf16 sP[KAG * CHUNK];           // 8 KB probabilities

  int tid = threadIdx.x, wave = tid >> 5;

  // TDM: DMA the whole contiguous K chunk (64x256 bf16 = 32 KB) into LDS.
  if (tid < 32) {
    tdm_load_to_lds(kbase, sKV, CHUNK * DIM);
    __builtin_amdgcn_s_wait_tensorcnt(0);
  }
  // gfx1250 prefetch: pull the V chunk toward cache while S1 is computed
  __builtin_prefetch(vbase + (size_t)tid * 64, 0, 1);
  __syncthreads();

  // S1 tiles: 4 m (agents) x 4 n (tokens)
  for (int t = wave; t < 16; t += 8) {
    int mt = t >> 2, nt = t & 3;
    v8f acc = vzero();
    for (int ks = 0; ks < DIM; ks += 32) {
      v16bf a = ld_a_bf(qa, DIM, mt * 16, ks);
      v16bf bfr = ld_bt_bf(sKV, DIM, nt * 16, ks);
      acc = wmma_bf16(a, bfr, acc);
    }
    st_c_f32(sS, CHUNK, mt * 16, nt * 16, acc, SCALE);
  }
  __syncthreads();

  // softmax over this chunk: 4 threads per agent row, 16 cols each
  {
    int row = tid >> 2, part = tid & 3;
    float m = -1e30f;
    for (int c = part * 16; c < part * 16 + 16; ++c)
      m = fmaxf(m, sS[row * CHUNK + c]);
    m = fmaxf(m, __shfl_xor(m, 1));
    m = fmaxf(m, __shfl_xor(m, 2));
    float l = 0.0f;
    for (int c = part * 16; c < part * 16 + 16; ++c) {
      float e = __expf(sS[row * CHUNK + c] - m);
      sP[row * CHUNK + c] = (__bf16)e;
      l += e;
    }
    l += __shfl_xor(l, 1);
    l += __shfl_xor(l, 2);
    if (part == 0) {
      size_t pi = ((size_t)b * NCH + chunk) * KAG + row;
      part_m[pi] = m;
      part_l[pi] = l;
    }
  }
  __syncthreads();

  {  // stage V chunk TRANSPOSED: sVt[d][token], ld = CHUNK
    __bf16* sVt = sKV;
    const v8bf* src = (const v8bf*)vbase;
    for (int i = tid; i < CHUNK * DIM / 8; i += 256) {
      v8bf vv = src[i];
      int tok = i / (DIM / 8);
      int d0 = (i % (DIM / 8)) * 8;
#pragma unroll
      for (int j = 0; j < 8; ++j) sVt[(d0 + j) * CHUNK + tok] = vv[j];
    }
  }
  __syncthreads();

  // acc = P1 @ Vc : 4 m x 16 n tiles, Kdim = CHUNK
  float* accout = part_acc + ((size_t)b * NCH + chunk) * KAG * DIM;
  for (int t = wave; t < 64; t += 8) {
    int mt = t & 3, nt = t >> 2;
    v8f acc = vzero();
    for (int ks = 0; ks < CHUNK; ks += 32) {
      v16bf a = ld_a_bf(sP, CHUNK, mt * 16, ks);
      v16bf bfr = ld_bt_bf(sKV, CHUNK, nt * 16, ks);  // sVt[d][token]
      acc = wmma_bf16(a, bfr, acc);
    }
    st_c_f32(accout, DIM, mt * 16, nt * 16, acc, 1.0f);
  }
}

// ---------------------------------------------------------------------------
// K4: merge stage-1 chunk partials (log-sum-exp) -> v1_raw bf16.
// ---------------------------------------------------------------------------
__global__ void stage1_reduce_kernel(const float* __restrict__ part_m,
                                     const float* __restrict__ part_l,
                                     const float* __restrict__ part_acc,
                                     __bf16* __restrict__ v1raw) {
  int row = blockIdx.x, b = blockIdx.y, tid = threadIdx.x;  // tid = d column
  float mstar = -1e30f;
  for (int s = 0; s < NCH; ++s)
    mstar = fmaxf(mstar, part_m[((size_t)b * NCH + s) * KAG + row]);
  float L = 0.0f;
  for (int s = 0; s < NCH; ++s) {
    size_t pi = ((size_t)b * NCH + s) * KAG + row;
    L += part_l[pi] * __expf(part_m[pi] - mstar);
  }
  float a = 0.0f;
  for (int s = 0; s < NCH; ++s) {
    size_t pi = ((size_t)b * NCH + s) * KAG + row;
    float w = __expf(part_m[pi] - mstar);
    a += part_acc[pi * DIM + tid] * w;
  }
  v1raw[((size_t)b * KAG + row) * DIM + tid] = (__bf16)(a / L);
}

// ---------------------------------------------------------------------------
// K5: v1 = v1_raw @ W_fc1^T + b_fc1, stored TRANSPOSED (v1t[d][agent]).
// ---------------------------------------------------------------------------
__global__ void fc1_kernel(const __bf16* __restrict__ v1raw,
                           const float* __restrict__ W,
                           const float* __restrict__ bias,
                           __bf16* __restrict__ v1t) {
  int b = blockIdx.x, tid = threadIdx.x, wave = tid >> 5;
  const __bf16* A = v1raw + (size_t)b * KAG * DIM;
  __bf16* outp = v1t + (size_t)b * DIM * KAG;
  for (int t = wave; t < 64; t += 8) {         // 4 m (agents) x 16 n (d)
    int mt = t & 3, nt = t >> 2;
    v8f acc = vzero();
    for (int ks = 0; ks < DIM; ks += 32) {
      v16bf a = ld_a_bf(A, DIM, mt * 16, ks);
      v16bf bfr = ld_bt_f32(W, DIM, nt * 16, ks);
      acc = wmma_bf16(a, bfr, acc);
    }
    int lane = tid & 31, hi = lane >> 4, nl = lane & 15;
    int col = nt * 16 + nl;                    // d
    float bv = bias[col];
#pragma unroll
    for (int r = 0; r < 8; ++r) {
      int row = mt * 16 + r + hi * 8;          // agent
      outp[(size_t)col * KAG + row] = (__bf16)(acc[r] + bv);
    }
  }
}

// ---------------------------------------------------------------------------
// K6: stage-2 + fc2 + residual: per 64-token tile.
// ---------------------------------------------------------------------------
__global__ void stage2_kernel(const __bf16* __restrict__ q,
                              const __bf16* __restrict__ ka,
                              const __bf16* __restrict__ v1t,
                              const float* __restrict__ Wfc2,
                              const float* __restrict__ bfc2,
                              const float* __restrict__ x,
                              float* __restrict__ out) {
  int tt = blockIdx.x, b = blockIdx.y;
  size_t rowbase = (size_t)b * NTOK + (size_t)tt * 64;
  const __bf16* Q = q + rowbase * DIM;

  __shared__ float  sS[64 * KAG];              // 16 KB
  __shared__ __bf16 sP[64 * KAG];              // 8 KB
  __shared__ __bf16 sV2[64 * DIM];             // 32 KB

  int tid = threadIdx.x, wave = tid >> 5;

  // S2: 4 m (tokens) x 4 n (agents)
  for (int t = wave; t < 16; t += 8) {
    int mt = t >> 2, nt = t & 3;
    v8f acc = vzero();
    for (int ks = 0; ks < DIM; ks += 32) {
      v16bf a = ld_a_bf(Q, DIM, mt * 16, ks);
      v16bf bfr = ld_bt_bf(ka, DIM, nt * 16, ks);
      acc = wmma_bf16(a, bfr, acc);
    }
    st_c_f32(sS, KAG, mt * 16, nt * 16, acc, SCALE);
  }
  __syncthreads();

  // exact softmax over the 64 agents (normalized probabilities into sP)
  {
    int row = tid >> 2, part = tid & 3;
    float m = -1e30f;
    for (int c = part * 16; c < part * 16 + 16; ++c)
      m = fmaxf(m, sS[row * KAG + c]);
    m = fmaxf(m, __shfl_xor(m, 1));
    m = fmaxf(m, __shfl_xor(m, 2));
    float l = 0.0f;
    float e16[16];
    for (int c = 0; c < 16; ++c) {
      e16[c] = __expf(sS[row * KAG + part * 16 + c] - m);
      l += e16[c];
    }
    l += __shfl_xor(l, 1);
    l += __shfl_xor(l, 2);
    float inv = 1.0f / l;
    for (int c = 0; c < 16; ++c)
      sP[row * KAG + part * 16 + c] = (__bf16)(e16[c] * inv);
  }
  __syncthreads();

  // v2 = P2 @ v1 : 4 m x 16 n, Kdim = 64 (v1 transposed -> contiguous loads)
  const __bf16* V1t = v1t + (size_t)b * DIM * KAG;
  for (int t = wave; t < 64; t += 8) {
    int mt = t & 3, nt = t >> 2;
    v8f acc = vzero();
    for (int ks = 0; ks < KAG; ks += 32) {
      v16bf a = ld_a_bf(sP, KAG, mt * 16, ks);
      v16bf bfr = ld_bt_bf(V1t, KAG, nt * 16, ks);
      acc = wmma_bf16(a, bfr, acc);
    }
    st_c_bf16(sV2, DIM, mt * 16, nt * 16, acc);
  }
  __syncthreads();

  // y = v2 @ Wfc2^T + b_fc2 + x  (pre-norm, f32)
  const float* xr = x + rowbase * DIM;
  float* outr = out + rowbase * DIM;
  for (int t = wave; t < 64; t += 8) {
    int mt = t & 3, nt = t >> 2;
    v8f acc = vzero();
    for (int ks = 0; ks < DIM; ks += 32) {
      v16bf a = ld_a_bf(sV2, DIM, mt * 16, ks);
      v16bf bfr = ld_bt_f32(Wfc2, DIM, nt * 16, ks);
      acc = wmma_bf16(a, bfr, acc);
    }
    int lane = tid & 31, hi = lane >> 4, nl = lane & 15;
    int col = nt * 16 + nl;
    float bv = bfc2[col];
#pragma unroll
    for (int r = 0; r < 8; ++r) {
      size_t row = (size_t)mt * 16 + r + hi * 8;
      outr[row * DIM + col] = acc[r] + bv + xr[row * DIM + col];
    }
  }
}

// ---------------------------------------------------------------------------
// K7: in-place RMSNorm over the last dim: out = scale * y / (rms + eps).
// ---------------------------------------------------------------------------
__global__ void rmsnorm_kernel(float* __restrict__ out,
                               const float* __restrict__ scale) {
  size_t row = blockIdx.x;
  float* p = out + row * DIM;
  int tid = threadIdx.x;
  float v = p[tid];
  float ss = v * v;
#pragma unroll
  for (int o = 16; o > 0; o >>= 1) ss += __shfl_xor(ss, o);
  __shared__ float wsum[8];
  __shared__ float stot;
  if ((tid & 31) == 0) wsum[tid >> 5] = ss;
  __syncthreads();
  if (tid == 0) {
    float s = 0.0f;
    for (int i = 0; i < 8; ++i) s += wsum[i];
    stot = s;
  }
  __syncthreads();
  float rms = sqrtf(stot * (1.0f / DIM));
  p[tid] = scale[tid] * (v / (rms + 1e-8f));
}

// ---------------------------------------------------------------------------
extern "C" void kernel_launch(void* const* d_in, const int* in_sizes, int n_in,
                              void* d_out, int out_size, void* d_ws, size_t ws_size,
                              hipStream_t stream) {
  (void)in_sizes; (void)n_in; (void)out_size; (void)ws_size;
  const float* agent = (const float*)d_in[0];
  const float* x     = (const float*)d_in[1];
  const float* Wqkv  = (const float*)d_in[2];
  const float* bqkv  = (const float*)d_in[3];
  const float* Wag   = (const float*)d_in[4];
  const float* bag   = (const float*)d_in[5];
  const float* Wfc1  = (const float*)d_in[6];
  const float* bfc1  = (const float*)d_in[7];
  const float* Wfc2  = (const float*)d_in[8];
  const float* bfc2  = (const float*)d_in[9];
  const float* nsc   = (const float*)d_in[10];
  float* out = (float*)d_out;

  char* ws = (char*)d_ws;
  size_t off = 0;
  auto alloc = [&](size_t bytes) -> void* {
    void* p = ws + off;
    off += (bytes + 255) & ~(size_t)255;
    return p;
  };

  __bf16* qa    = (__bf16*)alloc((size_t)KAG * DIM * 2);
  __bf16* ka    = (__bf16*)alloc((size_t)KAG * DIM * 2);
  __bf16* qb    = (__bf16*)alloc((size_t)BATCH * NTOK * DIM * 2);
  __bf16* kb    = (__bf16*)alloc((size_t)BATCH * NTOK * DIM * 2);
  __bf16* vb    = (__bf16*)alloc((size_t)BATCH * NTOK * DIM * 2);
  float*  pm    = (float*)alloc((size_t)BATCH * NCH * KAG * 4);
  float*  pl    = (float*)alloc((size_t)BATCH * NCH * KAG * 4);
  float*  pacc  = (float*)alloc((size_t)BATCH * NCH * KAG * DIM * 4);
  __bf16* v1raw = (__bf16*)alloc((size_t)BATCH * KAG * DIM * 2);
  __bf16* v1t   = (__bf16*)alloc((size_t)BATCH * DIM * KAG * 2);

  agent_proj_kernel<<<1, 256, 0, stream>>>(agent, Wag, bag, qa, ka);
  qkv_kernel<<<dim3((BATCH * NTOK) / 128, 12), 256, 0, stream>>>(x, Wqkv, bqkv,
                                                                 qb, kb, vb);
  stage1_kernel<<<dim3(NCH, BATCH), 256, 0, stream>>>(qa, kb, vb, pm, pl, pacc);
  stage1_reduce_kernel<<<dim3(KAG, BATCH), 256, 0, stream>>>(pm, pl, pacc, v1raw);
  fc1_kernel<<<BATCH, 256, 0, stream>>>(v1raw, Wfc1, bfc1, v1t);
  stage2_kernel<<<dim3(NTOK / 64, BATCH), 256, 0, stream>>>(qb, ka, v1t, Wfc2,
                                                            bfc2, x, out);
  rmsnorm_kernel<<<BATCH * NTOK, 256, 0, stream>>>(out, nsc);
}
